// SINKAttention_27814208209738
// MI455X (gfx1250) — matrix operand order repaired and checked by reading, not compiled, not run on hardware
//
#include <hip/hip_runtime.h>
#include <hip/hip_bf16.h>
#include <math.h>

// Problem constants (B=1)
#define SEQ 2048
#define EMB 768
#define NH  12
#define HD  64
#define EPS_SINK 1.0f
#define RCP_EPS  1.0f
#define LOG_MU  (-7.6246189861593985f)   // -log(2048)
#define LN_EPS  1e-12f

typedef __attribute__((ext_vector_type(16))) _Float16 v16h;
typedef __attribute__((ext_vector_type(8)))  _Float16 v8h;
typedef __attribute__((ext_vector_type(8)))  float    v8f;
typedef int v4i __attribute__((ext_vector_type(4)));

// ---------------------------------------------------------------------------
// Async global->LDS copy (CDNA5 GLOBAL_LOAD_ASYNC_TO_LDS_B128), with a
// synchronous 128-bit fallback if the builtin is not available.
// Builtin signature (from hipcc diagnostic): (v4i as1*, v4i as3*, imm, imm)
// ---------------------------------------------------------------------------
#if __has_builtin(__builtin_amdgcn_global_load_async_to_lds_b128) && \
    __has_builtin(__builtin_amdgcn_s_wait_asynccnt)
#define USE_ASYNC_LDS 1
#else
#define USE_ASYNC_LDS 0
#endif

typedef __attribute__((address_space(1))) v4i* gptr_b128;
typedef __attribute__((address_space(3))) v4i* lptr_b128;

__device__ __forceinline__ void cp_b128(_Float16* lds_dst, const _Float16* gsrc) {
#if USE_ASYNC_LDS
  __builtin_amdgcn_global_load_async_to_lds_b128(
      (gptr_b128)(gsrc), (lptr_b128)(lds_dst), 0, 0);
#else
  *(v8h*)lds_dst = *(const v8h*)gsrc;
#endif
}

__device__ __forceinline__ void wait_async0() {
#if USE_ASYNC_LDS
  __builtin_amdgcn_s_wait_asynccnt(0);
#endif
}

// ---------------------------------------------------------------------------
// WMMA fragment loads as contiguous 16B LDS vector loads (CDNA5 ISA §7.12.2)
// A tile in LDS: [m][k] row-major. Lane l: m=l&15, halves k=kh..kh+7 and
//   kh+16..kh+23 with kh=(l>>4)*8  -> two b128 loads.
// B tile in LDS: [n][k] (transposed). Lane l: n=l&15, halves k=kb..kb+7 and
//   kb+16..kb+23 with kb=(l>>4)*8 -> two b128 loads.
// ---------------------------------------------------------------------------
__device__ __forceinline__ v16h frag16(const _Float16* lo, const _Float16* hi) {
  v8h a = *(const v8h*)lo;
  v8h b = *(const v8h*)hi;
  return __builtin_shufflevector(a, b, 0, 1, 2, 3, 4, 5, 6, 7,
                                 8, 9, 10, 11, 12, 13, 14, 15);
}

template <int LD>
__device__ __forceinline__ v16h a_frag(const _Float16* tile, int lane) {
  const _Float16* row = tile + (lane & 15) * LD + (lane >> 4) * 8;
  return frag16(row, row + 16);
}

template <int LD>
__device__ __forceinline__ v16h b_frag(const _Float16* tile, int lane) {
  const _Float16* row = tile + (lane & 15) * LD + (lane >> 4) * 8;
  return frag16(row, row + 16);
}

// ---------------------------------------------------------------------------
// Reductions (wave32-aware)
// ---------------------------------------------------------------------------
__device__ __forceinline__ float block_sum256(float x) {
  #pragma unroll
  for (int off = 16; off > 0; off >>= 1) x += __shfl_xor(x, off, 32);
  __shared__ float red[8];
  int lane = threadIdx.x & 31, wave = threadIdx.x >> 5;
  if (lane == 0) red[wave] = x;
  __syncthreads();
  float t = red[0];
  #pragma unroll
  for (int w = 1; w < 8; ++w) t += red[w];
  __syncthreads();
  return t;
}

__device__ __forceinline__ void block_lse256(float& m, float& s) {
  #pragma unroll
  for (int off = 16; off > 0; off >>= 1) {
    float m2 = __shfl_xor(m, off, 32);
    float s2 = __shfl_xor(s, off, 32);
    if (m2 > m) { s = s * __expf(m - m2) + s2; m = m2; }
    else        { s = s + s2 * __expf(m2 - m); }
  }
  __shared__ float mred[8], sred[8];
  int lane = threadIdx.x & 31, wave = threadIdx.x >> 5;
  if (lane == 0) { mred[wave] = m; sred[wave] = s; }
  __syncthreads();
  float M = mred[0], S = sred[0];
  #pragma unroll
  for (int w = 1; w < 8; ++w) {
    float m2 = mred[w], s2 = sred[w];
    if (m2 > M) { S = S * __expf(M - m2) + s2; M = m2; }
    else        { S += s2 * __expf(m2 - M); }
  }
  m = M; s = S;
  __syncthreads();
}

// ---------------------------------------------------------------------------
// Pre-pass: f32 -> f16 convert, and transposed f32 -> f16 weight convert
// ---------------------------------------------------------------------------
__global__ __launch_bounds__(256) void convert_f16_kernel(
    const float* __restrict__ X, _Float16* __restrict__ Xh, int n) {
  int i = (blockIdx.x * 256 + threadIdx.x) * 4;
  if (i + 3 < n) {
    float4 x = *(const float4*)(X + i);
    Xh[i + 0] = (_Float16)x.x;
    Xh[i + 1] = (_Float16)x.y;
    Xh[i + 2] = (_Float16)x.z;
    Xh[i + 3] = (_Float16)x.w;
  }
}

// Wt[n][k] = (f16) W[k][n]   (W is [K][N] row-major)
__global__ __launch_bounds__(256) void transpose_f16_kernel(
    const float* __restrict__ W, _Float16* __restrict__ Wt, int K, int N) {
  __shared__ float tile[32][33];
  int k0 = blockIdx.x * 32, n0 = blockIdx.y * 32;
  int tx = threadIdx.x & 31, ty = threadIdx.x >> 5;   // 32 x 8
  #pragma unroll
  for (int i = 0; i < 32; i += 8)
    tile[ty + i][tx] = W[(size_t)(k0 + ty + i) * N + n0 + tx];
  __syncthreads();
  #pragma unroll
  for (int i = 0; i < 32; i += 8)
    Wt[(size_t)(n0 + ty + i) * K + k0 + tx] = (_Float16)tile[tx][ty + i];
}

__global__ void zero_kernel(float* __restrict__ p, int n) {
  int i = blockIdx.x * blockDim.x + threadIdx.x;
  if (i < n) p[i] = 0.0f;
}

// ---------------------------------------------------------------------------
// GEMM: C[M,N] = A[M,K] @ B + bias, A f16 [M][K], B given transposed f16
// Bt[N][K]. WG tile 64x128, 8 waves x 4 WMMA tiles, double-buffered async
// LDS staging. Epilogues:
//   MODE 0: Ch[row*N+col] = f16(acc+bias)
//   MODE 1: Ch[col*M+row] = f16(acc+bias)          (transposed store for Vt)
//   MODE 2: Cf[row*N+col] = acc+bias+resid         (f32 + residual)
// ---------------------------------------------------------------------------
template <int MODE>
__global__ __launch_bounds__(256) void gemm_f16_wmma_kernel(
    const _Float16* __restrict__ A, const _Float16* __restrict__ Bt,
    const float* __restrict__ bias, const float* __restrict__ resid,
    _Float16* __restrict__ Ch, float* __restrict__ Cf, int M, int N, int K)
{
  __shared__ alignas(16) _Float16 As[2][64][40];
  __shared__ alignas(16) _Float16 Bs[2][128][40];
  const int tile_m = blockIdx.x * 64;
  const int tile_n = blockIdx.y * 128;
  const int tid  = threadIdx.x;
  const int wave = tid >> 5, lane = tid & 31;
  const int wr = wave >> 1;            // 0..3 (16-row tile)
  const int wg = (wave & 1) * 4;       // col-tile group base (0 or 4)

  auto stage = [&](int buf, int k0) {
    {
      int r = tid >> 2, c = (tid & 3) * 8;             // 256 chunks: 64x32
      cp_b128(&As[buf][r][c], &A[(size_t)(tile_m + r) * K + k0 + c]);
    }
    #pragma unroll
    for (int t = 0; t < 2; ++t) {                      // 512 chunks: 128x32
      int ch = tid + t * 256;
      int r = ch >> 2, c = (ch & 3) * 8;
      cp_b128(&Bs[buf][r][c], &Bt[(size_t)(tile_n + r) * K + k0 + c]);
    }
  };

  v8f acc[4] = {};
  stage(0, 0);
  const int NK = K / 32;
  for (int ks = 0; ks < NK; ++ks) {
    const int b = ks & 1;
    wait_async0();
    __syncthreads();
    if (ks + 1 < NK) stage(b ^ 1, (ks + 1) * 32);
    v16h af = a_frag<40>(&As[b][wr * 16][0], lane);
    #pragma unroll
    for (int t = 0; t < 4; ++t) {
      v16h bf = b_frag<40>(&Bs[b][(wg + t) * 16][0], lane);
      acc[t] = __builtin_amdgcn_wmma_f32_16x16x32_f16(false, af, false, bf,
                                                      (short)0, acc[t],
                                                      false, false);
    }
    __syncthreads();
  }

  const int mm = (lane >> 4) * 8, nn = lane & 15;
  #pragma unroll
  for (int t = 0; t < 4; ++t) {
    const int col = tile_n + (wg + t) * 16 + nn;
    const float bcol = bias[col];
    #pragma unroll
    for (int r = 0; r < 8; ++r) {
      const int row = tile_m + wr * 16 + mm + r;
      float val = acc[t][r] + bcol;
      if (MODE == 0) {
        Ch[(size_t)row * N + col] = (_Float16)val;
      } else if (MODE == 1) {
        Ch[(size_t)col * M + row] = (_Float16)val;
      } else {
        Cf[(size_t)row * N + col] = val + resid[(size_t)row * N + col];
      }
    }
  }
}

// ---------------------------------------------------------------------------
// scores[h][i][j] = (Qh[i,h*64:] . Kh[j,h*64:]) / 8.  WG tile 64x128.
// Kh is [j][d] == [n][k] -> B tiles load straight, no transpose needed.
// ---------------------------------------------------------------------------
__global__ __launch_bounds__(256) void qk_scores_kernel(
    const _Float16* __restrict__ Qh, const _Float16* __restrict__ Kh,
    float* __restrict__ scores)
{
  __shared__ alignas(16) _Float16 Qs[64][72];
  __shared__ alignas(16) _Float16 Ks[128][72];
  const int h  = blockIdx.z;
  const int ti = blockIdx.x * 64;
  const int tj = blockIdx.y * 128;
  const int tid  = threadIdx.x;
  const int wave = tid >> 5, lane = tid & 31;
  const int wr = wave >> 1, wg = (wave & 1) * 4;

  #pragma unroll
  for (int t = 0; t < 2; ++t) {                        // 512 chunks: 64x64
    int ch = tid + t * 256;
    int r = ch >> 3, c = (ch & 7) * 8;
    cp_b128(&Qs[r][c], &Qh[(size_t)(ti + r) * EMB + h * HD + c]);
  }
  #pragma unroll
  for (int t = 0; t < 4; ++t) {                        // 1024 chunks: 128x64
    int ch = tid + t * 256;
    int r = ch >> 3, c = (ch & 7) * 8;
    cp_b128(&Ks[r][c], &Kh[(size_t)(tj + r) * EMB + h * HD + c]);
  }
  wait_async0();
  __syncthreads();

  v8f acc[4] = {};
  #pragma unroll
  for (int k0 = 0; k0 < 64; k0 += 32) {
    v16h af = a_frag<72>(&Qs[wr * 16][k0], lane);
    #pragma unroll
    for (int t = 0; t < 4; ++t) {
      v16h bf = b_frag<72>(&Ks[(wg + t) * 16][k0], lane);
      acc[t] = __builtin_amdgcn_wmma_f32_16x16x32_f16(false, af, false, bf,
                                                      (short)0, acc[t],
                                                      false, false);
    }
  }

  float* sh = scores + (size_t)h * SEQ * SEQ;
  const int mm = (lane >> 4) * 8, nn = lane & 15;
  #pragma unroll
  for (int t = 0; t < 4; ++t) {
    const int col = tj + (wg + t) * 16 + nn;
    #pragma unroll
    for (int r = 0; r < 8; ++r) {
      const int row = ti + wr * 16 + mm + r;
      sh[(size_t)row * SEQ + col] = acc[t][r] * 0.125f;
    }
  }
}

// ---------------------------------------------------------------------------
// Sinkhorn row / col updates: online one-pass logsumexp (f32, HBM-bound)
// ---------------------------------------------------------------------------
__global__ __launch_bounds__(256) void sink_row_kernel(
    const float* __restrict__ scores, float* __restrict__ u,
    const float* __restrict__ v)
{
  const int h = blockIdx.y;
  const int i = blockIdx.x;
  const float* row = scores + ((size_t)h * SEQ + i) * SEQ;
  const float* vh  = v + h * SEQ;
  const float  uu  = u[h * SEQ + i];

  float m = -INFINITY, s = 0.0f;
  for (int j = threadIdx.x; j < SEQ; j += 256) {
    float x = (row[j] + uu + vh[j]) * RCP_EPS;
    if (x > m) { s = s * __expf(m - x) + 1.0f; m = x; }
    else       { s += __expf(x - m); }
  }
  block_lse256(m, s);
  if (threadIdx.x == 0) {
    float lse = m + logf(s);
    u[h * SEQ + i] = EPS_SINK * (LOG_MU - lse) + uu;
  }
}

__global__ __launch_bounds__(256) void sink_col_kernel(
    const float* __restrict__ scores, const float* __restrict__ u,
    float* __restrict__ v)
{
  const int h = blockIdx.y;
  const int j = blockIdx.x * 256 + threadIdx.x;
  const float* sh = scores + (size_t)h * SEQ * SEQ;
  const float* uh = u + h * SEQ;
  const float  vv = v[h * SEQ + j];

  float m = -INFINITY, s = 0.0f;
  for (int i = 0; i < SEQ; ++i) {
    float x = (sh[(size_t)i * SEQ + j] + uh[i] + vv) * RCP_EPS;
    if (x > m) { s = s * __expf(m - x) + 1.0f; m = x; }
    else       { s += __expf(x - m); }
  }
  float lse = m + logf(s);
  v[h * SEQ + j] = EPS_SINK * (LOG_MU - lse) + vv;
}

// ---------------------------------------------------------------------------
// ctx[i,h*64+d] = sum_j exp((s+u+v)/eps)*n * V[j,h*64+d].  Vt is [E][S]
// (B-side [n=d][k=j] transposed layout) -> async B staging, fused exp A tile.
// ---------------------------------------------------------------------------
__global__ __launch_bounds__(256) void attn_v_kernel(
    const float* __restrict__ scores, const float* __restrict__ u,
    const float* __restrict__ v, const _Float16* __restrict__ Vt,
    _Float16* __restrict__ ctxh)
{
  __shared__ alignas(16) _Float16 Ats[64][40];   // attn tile [i][j]
  __shared__ alignas(16) _Float16 Vs[64][40];    // V tile [d][j]
  __shared__ float us[64];
  const int h  = blockIdx.y;
  const int ti = blockIdx.x * 64;
  const int tid  = threadIdx.x;
  const int wave = tid >> 5, lane = tid & 31;
  const int wr = wave >> 1, wc = (wave & 1) * 2;
  const float* sh = scores + (size_t)h * SEQ * SEQ;
  const float* vh = v + h * SEQ;

  if (tid < 64) us[tid] = u[h * SEQ + ti + tid];
  __syncthreads();

  v8f acc0 = {}, acc1 = {};
  for (int j0 = 0; j0 < SEQ; j0 += 32) {
    {
      int r = tid >> 2, c = (tid & 3) * 8;             // 256 chunks: 64x32
      cp_b128(&Vs[r][c], &Vt[(size_t)(h * HD + r) * SEQ + j0 + c]);
    }
    for (int i = tid; i < 64 * 32; i += 256) {
      int r = i >> 5, c = i & 31;
      float x = sh[(size_t)(ti + r) * SEQ + j0 + c];
      Ats[r][c] =
          (_Float16)(__expf((x + us[r] + vh[j0 + c]) * RCP_EPS) * (float)SEQ);
    }
    wait_async0();
    __syncthreads();

    v16h af  = a_frag<40>(&Ats[wr * 16][0], lane);
    v16h bf0 = b_frag<40>(&Vs[wc * 16][0], lane);
    v16h bf1 = b_frag<40>(&Vs[(wc + 1) * 16][0], lane);
    acc0 = __builtin_amdgcn_wmma_f32_16x16x32_f16(false, af, false, bf0,
                                                  (short)0, acc0, false, false);
    acc1 = __builtin_amdgcn_wmma_f32_16x16x32_f16(false, af, false, bf1,
                                                  (short)0, acc1, false, false);
    __syncthreads();
  }

  const int mm = (lane >> 4) * 8, nn = lane & 15;
  #pragma unroll
  for (int r = 0; r < 8; ++r) {
    int row = ti + wr * 16 + mm + r;
    ctxh[(size_t)row * EMB + h * HD + wc * 16 + nn]       = (_Float16)acc0[r];
    ctxh[(size_t)row * EMB + h * HD + (wc + 1) * 16 + nn] = (_Float16)acc1[r];
  }
}

// ---------------------------------------------------------------------------
// Row-wise LayerNorm over E=768 (3 elems/thread) -> final output
// ---------------------------------------------------------------------------
__global__ __launch_bounds__(256) void layernorm_kernel(
    const float* __restrict__ y, const float* __restrict__ g,
    const float* __restrict__ b, float* __restrict__ out)
{
  const int row = blockIdx.x;
  const float* yr = y + (size_t)row * EMB;
  float lx[3];
  float s = 0.0f;
  #pragma unroll
  for (int t = 0; t < 3; ++t) { lx[t] = yr[threadIdx.x + t * 256]; s += lx[t]; }
  float mu = block_sum256(s) * (1.0f / EMB);
  float vs = 0.0f;
  #pragma unroll
  for (int t = 0; t < 3; ++t) { float d = lx[t] - mu; vs += d * d; }
  float var = block_sum256(vs) * (1.0f / EMB);
  float inv = rsqrtf(var + LN_EPS);
  #pragma unroll
  for (int t = 0; t < 3; ++t) {
    int c = threadIdx.x + t * 256;
    out[(size_t)row * EMB + c] = g[c] * (lx[t] - mu) * inv + b[c];
  }
}

// ---------------------------------------------------------------------------
extern "C" void kernel_launch(void* const* d_in, const int* in_sizes, int n_in,
                              void* d_out, int out_size, void* d_ws, size_t ws_size,
                              hipStream_t stream) {
  (void)in_sizes; (void)n_in; (void)out_size; (void)ws_size;
  const float* X    = (const float*)d_in[0];
  const float* Wq   = (const float*)d_in[1];
  const float* bq   = (const float*)d_in[2];
  const float* Wk   = (const float*)d_in[3];
  const float* bk   = (const float*)d_in[4];
  const float* Wv   = (const float*)d_in[5];
  const float* bv   = (const float*)d_in[6];
  const float* Wo   = (const float*)d_in[7];
  const float* bo   = (const float*)d_in[8];
  const float* ln_g = (const float*)d_in[9];
  const float* ln_b = (const float*)d_in[10];
  float* out = (float*)d_out;

  const size_t SE = (size_t)SEQ * EMB;   // 1,572,864
  const size_t EE = (size_t)EMB * EMB;   //   589,824
  const size_t HS = (size_t)NH * SEQ;    //    24,576

  char* p = (char*)d_ws;
  auto alloc = [&](size_t bytes) -> char* {
    char* q = p;
    p += (bytes + 255) & ~(size_t)255;
    return q;
  };
  float*     scores = (float*)alloc(NH * (size_t)SEQ * SEQ * 4);
  float*     u      = (float*)alloc(HS * 4);
  float*     v      = (float*)alloc(HS * 4);
  float*     yb     = (float*)alloc(SE * 4);
  _Float16*  Xh     = (_Float16*)alloc(SE * 2);
  _Float16*  Wqt    = (_Float16*)alloc(EE * 2);
  _Float16*  Wkt    = (_Float16*)alloc(EE * 2);
  _Float16*  Wvt    = (_Float16*)alloc(EE * 2);
  _Float16*  Wot    = (_Float16*)alloc(EE * 2);
  _Float16*  Qh     = (_Float16*)alloc(SE * 2);
  _Float16*  Kh     = (_Float16*)alloc(SE * 2);
  _Float16*  Vt     = (_Float16*)alloc(SE * 2);
  _Float16*  ctxh   = (_Float16*)alloc(SE * 2);

  dim3 blk(256);

  // 0) pre-pass: f16 operands + transposed weights; zero u,v
  convert_f16_kernel<<<dim3((unsigned)(SE / 1024)), blk, 0, stream>>>(X, Xh, (int)SE);
  dim3 tr_grid(EMB / 32, EMB / 32);
  transpose_f16_kernel<<<tr_grid, blk, 0, stream>>>(Wq, Wqt, EMB, EMB);
  transpose_f16_kernel<<<tr_grid, blk, 0, stream>>>(Wk, Wkt, EMB, EMB);
  transpose_f16_kernel<<<tr_grid, blk, 0, stream>>>(Wv, Wvt, EMB, EMB);
  transpose_f16_kernel<<<tr_grid, blk, 0, stream>>>(Wo, Wot, EMB, EMB);
  zero_kernel<<<dim3((unsigned)((2 * HS + 255) / 256)), blk, 0, stream>>>(u, (int)(2 * HS));

  // 1) QKV projections (V written transposed as Vt[E][S])
  dim3 gemm_grid(SEQ / 64, EMB / 128);
  gemm_f16_wmma_kernel<0><<<gemm_grid, blk, 0, stream>>>(Xh, Wqt, bq, nullptr, Qh, nullptr, SEQ, EMB, EMB);
  gemm_f16_wmma_kernel<0><<<gemm_grid, blk, 0, stream>>>(Xh, Wkt, bk, nullptr, Kh, nullptr, SEQ, EMB, EMB);
  gemm_f16_wmma_kernel<1><<<gemm_grid, blk, 0, stream>>>(Xh, Wvt, bv, nullptr, Vt, nullptr, SEQ, EMB, EMB);

  // 2) scores = Q K^T / sqrt(D)
  dim3 sc_grid(SEQ / 64, SEQ / 128, NH);
  qk_scores_kernel<<<sc_grid, blk, 0, stream>>>(Qh, Kh, scores);

  // 3) Sinkhorn iterations
  for (int it = 0; it < 3; ++it) {
    sink_row_kernel<<<dim3(SEQ, NH), blk, 0, stream>>>(scores, u, v);
    sink_col_kernel<<<dim3(SEQ / 256, NH), blk, 0, stream>>>(scores, u, v);
  }

  // 4) ctx = (exp(plan) * n) @ V  -> f16
  attn_v_kernel<<<dim3(SEQ / 64, NH), blk, 0, stream>>>(scores, u, v, Vt, ctxh);

  // 5) y = ctx @ Wo + bo + X  (f32 + residual)
  gemm_f16_wmma_kernel<2><<<gemm_grid, blk, 0, stream>>>(ctxh, Wot, bo, X, nullptr, yb, SEQ, EMB, EMB);

  // 6) LayerNorm -> out
  layernorm_kernel<<<dim3(SEQ), blk, 0, stream>>>(yb, ln_g, ln_b, out);
}
